// EnhancedDecoderLayer_34359738368029
// MI455X (gfx1250) — matrix operand order repaired and checked by reading, not compiled, not run on hardware
//
#include <hip/hip_runtime.h>

// ---------------------------------------------------------------------------
// Types for gfx1250 WMMA (wave32): v_wmma_f32_16x16x32_bf16
// ---------------------------------------------------------------------------
typedef __bf16        v16bf __attribute__((ext_vector_type(16)));
typedef float         v8f   __attribute__((ext_vector_type(8)));
typedef unsigned int  v8ui  __attribute__((ext_vector_type(8)));

__device__ __forceinline__ unsigned short f2bf(float f) {
  unsigned int u = __builtin_bit_cast(unsigned int, f);
  u += 0x7fffu + ((u >> 16) & 1u);     // round-to-nearest-even
  return (unsigned short)(u >> 16);
}

__device__ __forceinline__ v8f wmma_bf16(v16bf a, v16bf b, v8f c) {
  // (neg_a, A, neg_b, B, c_mod, C, reuse_a, reuse_b)
  return __builtin_amdgcn_wmma_f32_16x16x32_bf16(false, a, false, b, (short)0, c,
                                                 false, false);
}

// A-matrix 16x32 bf16 fragment (ISA 7.12.2): lane m=l&15, half=l>>4,
// VGPR p: k = (p<4?0:16) + 8*half + 2*(p&3) (+1). rowp -> LDS row, K contiguous.
__device__ __forceinline__ v16bf load_frag_a(const unsigned short* rowp, int lane, int k0) {
  const int half = (lane >> 4) & 1;
  v8ui u;
#pragma unroll
  for (int p = 0; p < 8; ++p) {
    int kk = k0 + ((p < 4) ? 0 : 16) + 8 * half + 2 * (p & 3);
    u[p] = *(const unsigned int*)(rowp + kk);
  }
  return __builtin_bit_cast(v16bf, u);
}

// B-matrix 32x16 bf16 fragment: lane n=l&15, half=l>>4, VGPR p: k = (p<4?0:16) + 8*half + 2*(p&3).
// rowp points to LDS row for column n, K contiguous.
__device__ __forceinline__ v16bf load_frag_b(const unsigned short* rowp, int lane, int k0) {
  const int half = (lane >> 4) & 1;
  v8ui u;
#pragma unroll
  for (int p = 0; p < 8; ++p) {
    int kk = k0 + ((p < 4) ? 0 : 16) + 8 * half + 2 * (p & 3);
    u[p] = *(const unsigned int*)(rowp + kk);
  }
  return __builtin_bit_cast(v16bf, u);
}

// ---------------------------------------------------------------------------
// Generic WMMA GEMM: C[M,N] = epilogue(A[M,K] @ B[K,N])
//   BT=1: B stored as [N,K] (transposed weights, e.g. conv1_w)
//   EP: 0=none, 1=resid + scale*acc, 2=relu, 3=resid + acc
// Tile: 64(M) x 128(N), 256 threads = 8 waves (4x2), K-step 32.
// ---------------------------------------------------------------------------
template <int BT, int EP>
__global__ __launch_bounds__(256) void gemm_kernel(const float* __restrict__ A,
                                                   const float* __restrict__ Bm,
                                                   float* __restrict__ C,
                                                   int Ndim, int Kdim,
                                                   const float* __restrict__ resid,
                                                   const float* __restrict__ scale_ptr) {
  __shared__ unsigned short lds_a[64][34];
  __shared__ unsigned short lds_b[128][34];   // stored N-major: lds_b[n][k]
  const int tid  = threadIdx.x;
  const int lane = tid & 31;
  const int wave = tid >> 5;
  const int wm   = wave & 3;     // 4 waves along M (16 rows each)
  const int wn   = wave >> 2;    // 2 waves along N (64 cols each)
  const int l16  = lane & 15;
  const int half = (lane >> 4) & 1;
  const int m0 = blockIdx.y * 64;
  const int n0 = blockIdx.x * 128;

  const v8f vz = {0.f, 0.f, 0.f, 0.f, 0.f, 0.f, 0.f, 0.f};
  v8f acc[4];
#pragma unroll
  for (int i = 0; i < 4; ++i) acc[i] = vz;

  for (int k0 = 0; k0 < Kdim; k0 += 32) {
    __syncthreads();
    // stage A tile 64x32 (fp32 -> bf16)
    for (int e = tid; e < 64 * 32; e += 256) {
      int r = e >> 5, kk = e & 31;
      lds_a[r][kk] = f2bf(A[(size_t)(m0 + r) * Kdim + k0 + kk]);
    }
    // stage B tile 32x128 into N-major layout
    if (BT) {
      for (int e = tid; e < 32 * 128; e += 256) {
        int n = e >> 5, kk = e & 31;
        lds_b[n][kk] = f2bf(Bm[(size_t)(n0 + n) * Kdim + k0 + kk]);
      }
    } else {
      for (int e = tid; e < 32 * 128; e += 256) {
        int n = e & 127, kk = e >> 7;
        lds_b[n][kk] = f2bf(Bm[(size_t)(k0 + kk) * Ndim + n0 + n]);
      }
    }
    __syncthreads();
    v16bf a = load_frag_a(&lds_a[wm * 16 + l16][0], lane, 0);
#pragma unroll
    for (int nt = 0; nt < 4; ++nt) {
      v16bf bb = load_frag_b(&lds_b[wn * 64 + nt * 16 + l16][0], lane, 0);
      acc[nt] = wmma_bf16(a, bb, acc[nt]);
    }
  }

  const float sc = (EP == 1) ? scale_ptr[0] : 1.0f;
#pragma unroll
  for (int nt = 0; nt < 4; ++nt) {
#pragma unroll
    for (int r = 0; r < 8; ++r) {
      int m = m0 + wm * 16 + r + 8 * half;
      int n = n0 + wn * 64 + nt * 16 + l16;
      size_t idx = (size_t)m * Ndim + n;
      float v = acc[nt][r];
      if (EP == 1) v = resid[idx] + sc * v;
      else if (EP == 2) v = fmaxf(v, 0.f);
      else if (EP == 3) v = resid[idx] + v;
      C[idx] = v;
    }
  }
}

// ---------------------------------------------------------------------------
// Flash-style attention: per (b,h,query-tile of 64). dh=64, L=S=512, scale=1/8.
// 4 waves, each owns 16 query rows. Streaming softmax in C-layout:
// accumulator VGPR r of lane l <-> row r + 8*(l>>4), col = l&15 (+16*nt).
// ---------------------------------------------------------------------------
__global__ __launch_bounds__(128) void attn_kernel(const float* __restrict__ Q,
                                                   const float* __restrict__ K,
                                                   const float* __restrict__ V,
                                                   float* __restrict__ O) {
  __shared__ unsigned short q_lds[64][66];       // [qrow][dh]
  __shared__ unsigned short k_lds[64][66];       // [key][dh]   (B for S=Q@K^T)
  __shared__ unsigned short v_lds[64][66];       // [dh][key]   (B for O=P@V)
  __shared__ unsigned short p_lds[4][16][66];    // per-wave P tile [row][key]

  const int tid  = threadIdx.x;
  const int lane = tid & 31;
  const int wave = tid >> 5;
  const int l16  = lane & 15;
  const int half = (lane >> 4) & 1;
  const int qt = blockIdx.x, h = blockIdx.y, b = blockIdx.z;

  const size_t qbase = ((size_t)(b * 512 + qt * 64)) * 512 + h * 64;
  for (int e = tid; e < 64 * 64; e += 128) {
    int r = e >> 6, c = e & 63;
    q_lds[r][c] = f2bf(Q[qbase + (size_t)r * 512 + c] * 0.125f);  // 1/sqrt(64)
  }

  const v8f vz = {0.f, 0.f, 0.f, 0.f, 0.f, 0.f, 0.f, 0.f};
  v8f o_acc[4];
  float m_run[8], l_run[8];
#pragma unroll
  for (int i = 0; i < 4; ++i) o_acc[i] = vz;
#pragma unroll
  for (int r = 0; r < 8; ++r) { m_run[r] = -3.0e38f; l_run[r] = 0.f; }

  for (int kt = 0; kt < 8; ++kt) {
    __syncthreads();
    const size_t kvb = ((size_t)(b * 512 + kt * 64)) * 512 + h * 64;
    for (int e = tid; e < 64 * 64; e += 128) {
      int r = e >> 6, c = e & 63;
      k_lds[r][c] = f2bf(K[kvb + (size_t)r * 512 + c]);
      v_lds[c][r] = f2bf(V[kvb + (size_t)r * 512 + c]);   // transpose V
    }
    __syncthreads();

    // S tile: 16 rows x 64 keys per wave
    v8f s_acc[4];
#pragma unroll
    for (int nt = 0; nt < 4; ++nt) s_acc[nt] = vz;
#pragma unroll
    for (int ks = 0; ks < 2; ++ks) {
      v16bf a = load_frag_a(&q_lds[wave * 16 + l16][0], lane, ks * 32);
#pragma unroll
      for (int nt = 0; nt < 4; ++nt) {
        v16bf bb = load_frag_b(&k_lds[nt * 16 + l16][0], lane, ks * 32);
        s_acc[nt] = wmma_bf16(a, bb, s_acc[nt]);
      }
    }

    // streaming softmax (reduce across the 16-lane half owning each row)
#pragma unroll
    for (int r = 0; r < 8; ++r) {
      float tm = fmaxf(fmaxf(s_acc[0][r], s_acc[1][r]),
                       fmaxf(s_acc[2][r], s_acc[3][r]));
#pragma unroll
      for (int off = 8; off >= 1; off >>= 1) tm = fmaxf(tm, __shfl_xor(tm, off, 32));
      float nm   = fmaxf(m_run[r], tm);
      float corr = __expf(m_run[r] - nm);
      m_run[r]   = nm;
      float ps = 0.f;
#pragma unroll
      for (int nt = 0; nt < 4; ++nt) {
        float pv = __expf(s_acc[nt][r] - nm);
        s_acc[nt][r] = pv;
        ps += pv;
      }
#pragma unroll
      for (int off = 8; off >= 1; off >>= 1) ps += __shfl_xor(ps, off, 32);
      l_run[r] = l_run[r] * corr + ps;
#pragma unroll
      for (int nt = 0; nt < 4; ++nt) o_acc[nt][r] = o_acc[nt][r] * corr;
    }

    // P (C-layout) -> LDS -> A fragment
#pragma unroll
    for (int nt = 0; nt < 4; ++nt)
#pragma unroll
      for (int r = 0; r < 8; ++r)
        p_lds[wave][r + 8 * half][nt * 16 + l16] = f2bf(s_acc[nt][r]);
    __builtin_amdgcn_wave_barrier();   // per-wave LDS is in-order; fence scheduler

#pragma unroll
    for (int ks = 0; ks < 2; ++ks) {
      v16bf a = load_frag_a(&p_lds[wave][l16][0], lane, ks * 32);
#pragma unroll
      for (int nt = 0; nt < 4; ++nt) {
        v16bf bb = load_frag_b(&v_lds[nt * 16 + l16][0], lane, ks * 32);
        o_acc[nt] = wmma_bf16(a, bb, o_acc[nt]);
      }
    }
  }

#pragma unroll
  for (int nt = 0; nt < 4; ++nt)
#pragma unroll
    for (int r = 0; r < 8; ++r) {
      int row = qt * 64 + wave * 16 + r + 8 * half;
      O[((size_t)(b * 512 + row)) * 512 + h * 64 + nt * 16 + l16] =
          o_acc[nt][r] / l_run[r];
    }
}

// ---------------------------------------------------------------------------
// Decomposition helpers
// ---------------------------------------------------------------------------
__global__ void mean_kernel(const float* __restrict__ X, float* __restrict__ xg) {
  int t = blockIdx.x * blockDim.x + threadIdx.x;
  if (t >= 32 * 512) return;
  int b = t >> 9, d = t & 511;
  float s = 0.f;
  for (int l = 0; l < 512; ++l) s += X[((size_t)(b * 512 + l)) * 512 + d];
  xg[t] = s * (1.f / 512.f);
}

__global__ __launch_bounds__(256) void kp_kernel(const float* __restrict__ xg,
                                                 const float* __restrict__ w1,
                                                 const float* __restrict__ b1,
                                                 const float* __restrict__ w2,
                                                 const float* __restrict__ b2,
                                                 int* __restrict__ kb,
                                                 int* __restrict__ halfb) {
  const int b = blockIdx.x, j = threadIdx.x;
  __shared__ float red[256];
  float h = b1[j];
  for (int d = 0; d < 512; ++d) h += xg[b * 512 + d] * w1[d * 256 + j];
  h = fmaxf(h, 0.f);
  red[j] = h * w2[j];
  __syncthreads();
  for (int s = 128; s > 0; s >>= 1) {
    if (j < s) red[j] += red[j + s];
    __syncthreads();
  }
  if (j == 0) {
    float z   = red[0] + b2[0];
    float sig = 1.f / (1.f + __expf(-z));
    int k = (int)rintf(sig * 45.f + 5.f);
    k = (k > 50) ? 50 : k;
    k = (k < 3) ? 3 : k;
    if ((k & 1) == 0) k -= 1;
    k = (k < 3) ? 3 : k;
    kb[b] = k;
    halfb[b] = k >> 1;
  }
}

__global__ void softw_kernel(const float* __restrict__ dwS, const int* __restrict__ kb,
                             float* __restrict__ wbuf) {
  int t = blockIdx.x * blockDim.x + threadIdx.x;
  if (t >= 32 * 512) return;
  int b = t >> 9, d = t & 511;
  int k = kb[b];
  float m = -3.0e38f;
  for (int j = 0; j < k; ++j) m = fmaxf(m, dwS[d * 50 + j]);
  float s = 0.f;
  for (int j = 0; j < k; ++j) s += __expf(dwS[d * 50 + j] - m);
  float inv = 1.f / s;
  for (int j = 0; j < 50; ++j)
    wbuf[((size_t)t) * 50 + j] = (j < k) ? __expf(dwS[d * 50 + j] - m) * inv : 0.f;
}

__global__ void trendconv_kernel(const float* __restrict__ Xin,
                                 const float* __restrict__ wbuf,
                                 const int* __restrict__ halfb,
                                 float* __restrict__ Xout,
                                 float* __restrict__ Tsum, int first) {
  size_t t = (size_t)blockIdx.x * 256 + threadIdx.x;
  if (t >= (size_t)32 * 512 * 512) return;
  int d = (int)(t & 511);
  int l = (int)((t >> 9) & 511);
  int b = (int)(t >> 18);
  int hf = halfb[b];
  const float* wrow = wbuf + ((size_t)(b * 512 + d)) * 50;
  float acc = 0.f;
  for (int j = 0; j < 50; ++j) {
    int idx = l - hf + j;
    idx = idx < 0 ? 0 : (idx > 511 ? 511 : idx);   // edge replicate pad
    acc += wrow[j] * Xin[((size_t)(b * 512 + idx)) * 512 + d];
  }
  Xout[t] = Xin[t] - acc;
  Tsum[t] = first ? acc : (Tsum[t] + acc);
}

// t_cat[b,l, i*3+dt] = t_sum[b, wrap(l-1+dt), i]  (circular conv as GEMM)
__global__ void tcat_kernel(const float* __restrict__ Tsum, float* __restrict__ tcat) {
  size_t t = (size_t)blockIdx.x * 256 + threadIdx.x;
  if (t >= (size_t)32 * 512 * 512) return;
  int i = (int)(t & 511);
  int l = (int)((t >> 9) & 511);
  int b = (int)(t >> 18);
  for (int dt = 0; dt < 3; ++dt) {
    int lp = (l - 1 + dt + 512) & 511;
    tcat[((size_t)(b * 512 + l)) * 1536 + i * 3 + dt] =
        Tsum[((size_t)(b * 512 + lp)) * 512 + i];
  }
}

// Bp[(i*3+dt)*512 + o] = proj_w1[o,i,dt]
__global__ void packw1_kernel(const float* __restrict__ w1, float* __restrict__ Bp) {
  int t = blockIdx.x * blockDim.x + threadIdx.x;
  if (t >= 512 * 512 * 3) return;
  int o  = t % 512;
  int i  = (t / 512) % 512;
  int dt = t / (512 * 512);
  Bp[((size_t)(i * 3 + dt)) * 512 + o] = w1[((size_t)(o * 512 + i)) * 3 + dt];
}

// ---------------------------------------------------------------------------
// Launch: B=32, L=S=512, D=512, H=8, DFF=2048, COUT=512, KMAX=50, KP_H=256
// ---------------------------------------------------------------------------
extern "C" void kernel_launch(void* const* d_in, const int* in_sizes, int n_in,
                              void* d_out, int out_size, void* d_ws, size_t ws_size,
                              hipStream_t stream) {
  (void)in_sizes; (void)n_in; (void)out_size; (void)ws_size;
  const float* x       = (const float*)d_in[0];
  const float* cross   = (const float*)d_in[1];
  const float* wq_s    = (const float*)d_in[2];
  const float* wk_s    = (const float*)d_in[3];
  const float* wv_s    = (const float*)d_in[4];
  const float* wo_s    = (const float*)d_in[5];
  const float* wq_c    = (const float*)d_in[6];
  const float* wk_c    = (const float*)d_in[7];
  const float* wv_c    = (const float*)d_in[8];
  const float* wo_c    = (const float*)d_in[9];
  const float* conv1_w = (const float*)d_in[10];
  const float* conv2_w = (const float*)d_in[11];
  const float* dw      = (const float*)d_in[12];
  const float* kp_w1   = (const float*)d_in[13];
  const float* kp_b1   = (const float*)d_in[14];
  const float* kp_w2   = (const float*)d_in[15];
  const float* kp_b2   = (const float*)d_in[16];
  const float* proj_w1 = (const float*)d_in[17];
  const float* proj_w2 = (const float*)d_in[18];
  const float* s_scale = (const float*)d_in[19];
  const float* c_scale = (const float*)d_in[20];

  const size_t M   = 16384;                 // B*L
  const size_t SZX = M * 512 * sizeof(float);
  char* ws = (char*)d_ws;
  float* qb   = (float*)(ws + 0 * SZX);
  float* kbuf = (float*)(ws + 1 * SZX);
  float* vbuf = (float*)(ws + 2 * SZX);
  float* ao   = (float*)(ws + 3 * SZX);
  float* xA   = (float*)(ws + 4 * SZX);
  float* xB   = (float*)(ws + 5 * SZX);
  float* tsum = (float*)(ws + 6 * SZX);
  float* hbuf = (float*)(ws + 7 * SZX);           // 128 MB: FFN hidden / t_cat
  char*  p = ws + 7 * SZX + M * 2048 * sizeof(float);
  float* wtap = (float*)p; p += (size_t)32 * 512 * 50 * sizeof(float);
  float* xg   = (float*)p; p += 32 * 512 * sizeof(float);
  int*   kb   = (int*)p;   p += 128;
  int*   hb   = (int*)p;   p += 128;
  float* Bp   = (float*)p;
  float* xout = (float*)d_out;
  float* tout = (float*)d_out + M * 512;
  float* tcat  = hbuf;
  float* hconv = qb;   // free by trend-projection time

  const dim3 blk(256);
  const dim3 g512(512 / 128, M / 64);
  const dim3 g2048(2048 / 128, M / 64);
  const dim3 agrid(8, 8, 32);

  // ---- self attention ----
  gemm_kernel<0,0><<<g512, blk, 0, stream>>>(x, wq_s, qb,   512, 512, nullptr, nullptr);
  gemm_kernel<0,0><<<g512, blk, 0, stream>>>(x, wk_s, kbuf, 512, 512, nullptr, nullptr);
  gemm_kernel<0,0><<<g512, blk, 0, stream>>>(x, wv_s, vbuf, 512, 512, nullptr, nullptr);
  attn_kernel<<<agrid, dim3(128), 0, stream>>>(qb, kbuf, vbuf, ao);
  gemm_kernel<0,1><<<g512, blk, 0, stream>>>(ao, wo_s, xA, 512, 512, x, s_scale);
  // ---- decomp 0 ----
  mean_kernel<<<64, 256, 0, stream>>>(xA, xg);
  kp_kernel<<<32, 256, 0, stream>>>(xg, kp_w1, kp_b1, kp_w2, kp_b2, kb, hb);
  softw_kernel<<<64, 256, 0, stream>>>(dw, kb, wtap);
  trendconv_kernel<<<32768, 256, 0, stream>>>(xA, wtap, hb, xB, tsum, 1);
  // ---- cross attention ----
  gemm_kernel<0,0><<<g512, blk, 0, stream>>>(xB,    wq_c, qb,   512, 512, nullptr, nullptr);
  gemm_kernel<0,0><<<g512, blk, 0, stream>>>(cross, wk_c, kbuf, 512, 512, nullptr, nullptr);
  gemm_kernel<0,0><<<g512, blk, 0, stream>>>(cross, wv_c, vbuf, 512, 512, nullptr, nullptr);
  attn_kernel<<<agrid, dim3(128), 0, stream>>>(qb, kbuf, vbuf, ao);
  gemm_kernel<0,1><<<g512, blk, 0, stream>>>(ao, wo_c, xA, 512, 512, xB, c_scale);
  // ---- decomp 1 ----
  mean_kernel<<<64, 256, 0, stream>>>(xA, xg);
  kp_kernel<<<32, 256, 0, stream>>>(xg, kp_w1 + 512 * 256, kp_b1 + 256,
                                    kp_w2 + 256, kp_b2 + 1, kb, hb);
  softw_kernel<<<64, 256, 0, stream>>>(dw + 512 * 50, kb, wtap);
  trendconv_kernel<<<32768, 256, 0, stream>>>(xA, wtap, hb, xB, tsum, 0);
  // ---- conv FFN ----
  gemm_kernel<1,2><<<g2048, blk, 0, stream>>>(xB, conv1_w, hbuf, 2048, 512, nullptr, nullptr);
  gemm_kernel<1,3><<<g512, blk, 0, stream>>>(hbuf, conv2_w, xA, 512, 2048, xB, nullptr);
  // ---- decomp 2 (seasonal -> first half of d_out) ----
  mean_kernel<<<64, 256, 0, stream>>>(xA, xg);
  kp_kernel<<<32, 256, 0, stream>>>(xg, kp_w1 + 2 * 512 * 256, kp_b1 + 2 * 256,
                                    kp_w2 + 2 * 256, kp_b2 + 2, kb, hb);
  softw_kernel<<<64, 256, 0, stream>>>(dw + 2 * 512 * 50, kb, wtap);
  trendconv_kernel<<<32768, 256, 0, stream>>>(xA, wtap, hb, xout, tsum, 0);
  // ---- trend projection (circular k=3 conv as GEMM, relu, 1x1 conv) ----
  tcat_kernel<<<32768, 256, 0, stream>>>(tsum, tcat);
  packw1_kernel<<<(512 * 512 * 3 + 255) / 256, 256, 0, stream>>>(proj_w1, Bp);
  gemm_kernel<0,2><<<g512, blk, 0, stream>>>(tcat, Bp, hconv, 512, 1536, nullptr, nullptr);
  gemm_kernel<1,0><<<g512, blk, 0, stream>>>(hconv, proj_w2, tout, 512, 512, nullptr, nullptr);
}
